// GridSampleConv_66451734004048
// MI455X (gfx1250) — matrix operand order repaired and checked by reading, not compiled, not run on hardware
//
#include <hip/hip_runtime.h>

// ---------------- problem constants ----------------
#define NQ 50000          // query points
#define NS 100000         // source points
#define KNB 32            // neighbors per query
#define KK 27             // kernel points
#define CH 64             // feature channels (in == out)
#define KI 1728           // KK*CH, contraction length of stage 2 (= 54 * 32)
#define NEG_SLOPE 0.01f
#define KP_EXT 0.05773502691896258f   // 2*0.1/2/sqrt(3)

typedef __attribute__((ext_vector_type(16))) __bf16 v16bf;
typedef __attribute__((ext_vector_type(8)))  __bf16 v8bf;
typedef __attribute__((ext_vector_type(8)))  float  v8f;
typedef __attribute__((ext_vector_type(4)))  float  v4f;

static __device__ __forceinline__ v8f wmma_bf16(v16bf a, v16bf b, v8f c) {
  // D(f32 16x16) = A(bf16 16x32) * B(bf16 32x16) + C
  return __builtin_amdgcn_wmma_f32_16x16x32_bf16(false, a, false, b, (short)0, c,
                                                 false, false);
}

// A-fragment K offset for lane-half h, slot j (16-bit A 16x32 layout, ISA 7.12.2)
static __device__ __forceinline__ int a_koff(int h, int j) {
  return (j < 8) ? ((h ? 8 : 0) + j) : ((h ? 24 : 16) + (j - 8));
}

// ---------------------------------------------------------------------------
// Kernel 0: repack kp_weights [27][64][64] f32 -> kpwT [64(o)][1728(l*64+i)] bf16
// ---------------------------------------------------------------------------
__global__ void kpw_pack_kernel(const float* __restrict__ kpw,
                                __bf16* __restrict__ kpwT) {
  int idx = blockIdx.x * blockDim.x + threadIdx.x;
  if (idx >= CH * KI) return;
  int o = idx / KI;
  int e = idx - o * KI;          // e = l*64 + i
  int l = e >> 6;
  int i = e & 63;
  kpwT[idx] = (__bf16)kpw[(l * CH + i) * CH + o];
}

// ---------------------------------------------------------------------------
// Kernel 1: feats = leaky_relu(x @ W_pre + b) -> bf16 [NS+1][64], shadow row = 0
// ---------------------------------------------------------------------------
__global__ void __launch_bounds__(128) preact_kernel(
    const float* __restrict__ x, const float* __restrict__ Wpre,
    const float* __restrict__ bpre, __bf16* __restrict__ featsB) {
  __shared__ __bf16 wT[CH][CH];   // [o][in] bf16 (transposed for B fragments)
  __shared__ float bs[CH];

  const int tid = threadIdx.x;
  for (int t = tid; t < CH * CH; t += 128) {
    int o = t & 63, in = t >> 6;                 // coalesced global read
    wT[o][in] = (__bf16)Wpre[in * CH + o];
  }
  if (tid < CH) bs[tid] = bpre[tid];
  __syncthreads();

  const int wv = tid >> 5, lane = tid & 31;
  const int lrow = lane & 15, half = lane >> 4;
  const int rowBase = blockIdx.x * 64 + wv * 16;
  const int row = rowBase + lrow;
  const bool interior = (int)(blockIdx.x + 1) * 64 <= NS;  // block-uniform

  v16bf a[2];
  if (interior) {
    const float* __restrict__ xr = x + row * CH;
#pragma unroll
    for (int c = 0; c < 2; ++c) {
      const int b0 = c * 32 + (half ? 8 : 0);        // slots 0..7
      const int b1 = b0 + 16;                        // slots 8..15
      v4f p0 = *(const v4f*)(xr + b0);
      v4f p1 = *(const v4f*)(xr + b0 + 4);
      v4f p2 = *(const v4f*)(xr + b1);
      v4f p3 = *(const v4f*)(xr + b1 + 4);
#pragma unroll
      for (int j = 0; j < 4; ++j) {
        a[c][j]      = (__bf16)p0[j];
        a[c][j + 4]  = (__bf16)p1[j];
        a[c][j + 8]  = (__bf16)p2[j];
        a[c][j + 12] = (__bf16)p3[j];
      }
    }
  } else {
#pragma unroll
    for (int c = 0; c < 2; ++c) {
#pragma unroll 1
      for (int j = 0; j < 16; ++j) {
        int k = c * 32 + a_koff(half, j);
        float v = (row < NS) ? x[row * CH + k] : 0.0f;
        a[c][j] = (__bf16)v;
      }
    }
  }

#pragma unroll
  for (int nt = 0; nt < 4; ++nt) {
    const int o = nt * 16 + lrow;
    v8f acc = {0.f, 0.f, 0.f, 0.f, 0.f, 0.f, 0.f, 0.f};
#pragma unroll
    for (int c = 0; c < 2; ++c) {
      v16bf b;
#pragma unroll
      for (int j = 0; j < 16; ++j) b[j] = wT[o][c * 32 + a_koff(half, j)];
      acc = wmma_bf16(a[c], b, acc);
    }
    const float bias = bs[o];
    if (interior) {
#pragma unroll
      for (int r = 0; r < 8; ++r) {
        int m = rowBase + r + half * 8;
        float v = acc[r] + bias;
        v = v > 0.f ? v : NEG_SLOPE * v;
        featsB[m * CH + o] = (__bf16)v;
      }
    } else {
#pragma unroll 1
      for (int r = 0; r < 8; ++r) {
        int m = rowBase + r + half * 8;
        if (m <= NS) {
          float v = acc[r] + bias;
          v = v > 0.f ? v : NEG_SLOPE * v;
          featsB[m * CH + o] = (m < NS) ? (__bf16)v : (__bf16)0.f;  // shadow = 0
        }
      }
    }
  }
}

// ---------------------------------------------------------------------------
// Kernel 2: KPConv. One block = 16 queries, 128 threads = 4 waves.
//  gather : 512 neighbor rows -> LDS via GLOBAL_LOAD_ASYNC_TO_LDS_B128
//  stage 1 (per query): g[l,i] = sum_k w[k,l]*nx[k,i]  -> 8 WMMA / query
//  stage 2 (per block): fx[16,64] = G[16,1728] @ kpwT   -> 54 WMMA / wave
// LDS: 64KB nx + 54KB G + ~8.7KB staging  (CDNA5: up to 320KB/WG)
// ---------------------------------------------------------------------------
__global__ void __launch_bounds__(128) kpconv_kernel(
    const float* __restrict__ q_pts, const float* __restrict__ s_pts,
    const long long* __restrict__ inds64, const float* __restrict__ kpts,
    const __bf16* __restrict__ featsB, const __bf16* __restrict__ kpwT,
    float* __restrict__ out) {
  __shared__ float  sq[16][3];          // query points
  __shared__ float  sp[16][KNB][3];     // neighbor - query (relative positions)
  __shared__ int    si[16][KNB];        // neighbor indices (shadow = NS)
  __shared__ float  skp[KK][3];         // kernel points
  __shared__ __bf16 nx[16][KNB][CH];    // gathered neighbor features, 64 KB
  __shared__ __bf16 G[16][KI];          // stage-1 output, 54 KB

  const int tid = threadIdx.x;
  const int qb = blockIdx.x;            // 3125 blocks * 16 queries

  // ---- index / point staging ----
  for (int t = tid; t < 16 * 3; t += 128)
    sq[t / 3][t % 3] = q_pts[(qb * 16 + t / 3) * 3 + (t % 3)];
  for (int t = tid; t < KK * 3; t += 128) skp[t / 3][t % 3] = kpts[t];
  for (int t = tid; t < 16 * KNB; t += 128) {
    long long raw = inds64[qb * (16 * KNB) + t];
    int m = (int)(raw % (long long)(NS + 1));
    if (m < 0) m += (NS + 1);           // python-style mod (-1 -> shadow)
    si[t / KNB][t % KNB] = m;
  }
  __syncthreads();

  // ---- async gather: 4096 x 16B segments (512 rows x 128B) into nx ----
  {
    const unsigned ldsBase = (unsigned)(size_t)&nx[0][0][0];  // LDS offset
#pragma unroll 4
    for (int it = 0; it < 32; ++it) {
      const int task = it * 128 + tid;        // segment id 0..4095
      const int n = task >> 3, seg = task & 7;
      const int ind = si[n >> 5][n & 31];
      const unsigned goff = (unsigned)ind * 128u + (unsigned)seg * 16u;
      const unsigned loff = ldsBase + (unsigned)task * 16u;
      asm volatile("global_load_async_to_lds_b128 %0, %1, %2"
                   :: "v"(loff), "v"(goff), "s"(featsB)
                   : "memory");
    }
  }

  // ---- overlap: relative neighbor positions while gathers are in flight ----
  for (int t = tid; t < 16 * KNB; t += 128) {
    int qq = t / KNB, kk = t % KNB, ind = si[qq][kk];
    float px, py, pz;
    if (ind < NS) {
      px = s_pts[ind * 3 + 0]; py = s_pts[ind * 3 + 1]; pz = s_pts[ind * 3 + 2];
    } else {
      px = py = pz = 1e6f;              // shadow point
    }
    sp[qq][kk][0] = px - sq[qq][0];
    sp[qq][kk][1] = py - sq[qq][1];
    sp[qq][kk][2] = pz - sq[qq][2];
  }
  asm volatile("s_wait_asynccnt 0" ::: "memory");
  __syncthreads();

  const int wv = tid >> 5, lane = tid & 31;
  const int lrow = lane & 15, half = lane >> 4;

  // ---- stage 1: per-query  g = w^T(32x32-padded l x k) @ nx(32 x 64) ----
  for (int qi = 0; qi < 4; ++qi) {
    const int q = wv * 4 + qi;
    // A fragments: KP correlation weights computed directly into registers
    v16bf a[2];
#pragma unroll
    for (int mt = 0; mt < 2; ++mt) {
      const int l = mt * 16 + lrow;
      const bool lv = (l < KK);
      const int ls = lv ? l : 0;
      const float kx = skp[ls][0], ky = skp[ls][1], kz = skp[ls][2];
#pragma unroll
      for (int j = 0; j < 16; ++j) {
        const int k = a_koff(half, j);
        float dx = sp[q][k][0] - kx;
        float dy = sp[q][k][1] - ky;
        float dz = sp[q][k][2] - kz;
        float d = sqrtf(dx * dx + dy * dy + dz * dz);
        float w = 1.f - d * (1.f / KP_EXT);
        w = w > 0.f ? w : 0.f;
        a[mt][j] = lv ? (__bf16)w : (__bf16)0.f;
      }
    }
#pragma unroll
    for (int nt = 0; nt < 4; ++nt) {
      const int i = nt * 16 + lrow;
      v16bf b;  // B fragment from LDS-resident gathered features
#pragma unroll
      for (int j = 0; j < 16; ++j) b[j] = nx[q][a_koff(half, j)][i];
#pragma unroll
      for (int mt = 0; mt < 2; ++mt) {
        v8f c = {0.f, 0.f, 0.f, 0.f, 0.f, 0.f, 0.f, 0.f};
        c = wmma_bf16(a[mt], b, c);
#pragma unroll
        for (int r = 0; r < 8; ++r) {
          const int l = mt * 16 + r + half * 8;
          if (l < KK) G[q][l * CH + i] = (__bf16)c[r];
        }
      }
    }
  }
  __syncthreads();

  // ---- stage 2: fx[16,64] = G[16,1728] @ kpw[1728,64]; wave wv owns cols ----
  {
    const int nt = wv;
    const int o = nt * 16 + lrow;
    const __bf16* __restrict__ bcol = kpwT + o * KI + (half ? 16 : 0);
    v8f acc = {0.f, 0.f, 0.f, 0.f, 0.f, 0.f, 0.f, 0.f};
#pragma unroll 2
    for (int c = 0; c < 54; ++c) {
      if (c + 8 < 54)  // stream-ahead hint -> global_prefetch_b8
        __builtin_prefetch(bcol + (c + 8) * 32, 0, 1);
      // A from LDS: two contiguous 8-element groups per lane
      v8bf lo = *(const v8bf*)&G[lrow][c * 32 + (half ? 16 : 0)];
      v8bf hi = *(const v8bf*)&G[lrow][c * 32 + 8 + (half ? 16 : 0)];
      v16bf a;
#pragma unroll
      for (int j = 0; j < 8; ++j) { a[j] = lo[j]; a[j + 8] = hi[j]; }
      // B from global: 16 contiguous bf16 (32B vector load, L2-resident)
      v16bf b = *(const v16bf*)(bcol + c * 32);
      acc = wmma_bf16(a, b, acc);
    }
#pragma unroll
    for (int r = 0; r < 8; ++r) {
      const int n = r + half * 8;
      out[(qb * 16 + n) * CH + o] = acc[r];
    }
  }
}

// ---------------------------------------------------------------------------
extern "C" void kernel_launch(void* const* d_in, const int* in_sizes, int n_in,
                              void* d_out, int out_size, void* d_ws,
                              size_t ws_size, hipStream_t stream) {
  (void)in_sizes; (void)n_in; (void)out_size; (void)ws_size;
  const float*     q_pts = (const float*)d_in[0];
  const float*     s_pts = (const float*)d_in[1];
  const long long* inds  = (const long long*)d_in[2];
  const float*     x     = (const float*)d_in[3];
  const float*     Wpre  = (const float*)d_in[4];
  const float*     bpre  = (const float*)d_in[5];
  const float*     kpw   = (const float*)d_in[6];
  const float*     kpts  = (const float*)d_in[7];
  float*           out   = (float*)d_out;

  // workspace layout: bf16 feats table [NS+1][64], then packed kp weights
  __bf16* featsB = (__bf16*)d_ws;
  size_t featsBytes = (size_t)(NS + 1) * CH * sizeof(__bf16);
  size_t off = (featsBytes + 255) & ~(size_t)255;
  __bf16* kpwT = (__bf16*)((char*)d_ws + off);

  kpw_pack_kernel<<<(CH * KI + 255) / 256, 256, 0, stream>>>(kpw, kpwT);
  preact_kernel<<<(NS + 1 + 63) / 64, 128, 0, stream>>>(x, Wpre, bpre, featsB);
  kpconv_kernel<<<NQ / 16, 128, 0, stream>>>(q_pts, s_pts, inds, kpts, featsB,
                                             kpwT, out);
}
